// CausalSelfAttention_82764019793932
// MI455X (gfx1250) — matrix operand order repaired and checked by reading, not compiled, not run on hardware
//
#include <hip/hip_runtime.h>
#include <hip/hip_bf16.h>

typedef __bf16 bf16;
typedef __attribute__((ext_vector_type(16))) __bf16 v16bf;
typedef __attribute__((ext_vector_type(8)))  __bf16 v8bf;
typedef __attribute__((ext_vector_type(8)))  float  v8f;

#define BATCH 8
#define SEQ   1024
#define EMBD  768
#define NHEAD 12
#define HDIM  64
#define BT    (BATCH*SEQ)   /* 8192 tokens */

static __device__ __forceinline__ v16bf cat8(v8bf lo, v8bf hi) {
  return __builtin_shufflevector(lo, hi, 0,1,2,3,4,5,6,7,8,9,10,11,12,13,14,15);
}
static __device__ __forceinline__ v8f wmma_bf16(v16bf a, v16bf b, v8f c) {
  // D(f32 16x16) = A(16x32 bf16) * B(32x16 bf16) + C
  return __builtin_amdgcn_wmma_f32_16x16x32_bf16(false, a, false, b, (short)0, c, false, false);
}

// ---------------------------------------------------------------------------
// f32 -> bf16 conversion (grid-stride)
// ---------------------------------------------------------------------------
__global__ __launch_bounds__(256)
void csa_cvt_bf16(const float* __restrict__ src, bf16* __restrict__ dst, int n) {
  for (int i = blockIdx.x * 256 + threadIdx.x; i < n; i += gridDim.x * 256)
    dst[i] = (bf16)src[i];
}

// ---------------------------------------------------------------------------
// GEMM: out = A[BT x EMBD] * W[EMBD x EMBD]^T + bias.
// One wave -> 32x64 output tile (2 M-subtiles x 4 N-subtiles, 8 accumulators).
// Per K-step: 4 A-frag loads + 8 B-frag loads feed 8 WMMAs (1.5 loads/WMMA).
// mode 0: write bf16 in [B,H,T,D] layout (QKV).  mode 1: write f32 row-major.
// ---------------------------------------------------------------------------
__global__ __launch_bounds__(256)
void csa_gemm_bf16(const bf16* __restrict__ A,
                   const bf16* __restrict__ W,
                   const float* __restrict__ bias,
                   bf16* __restrict__ out_bhtd,
                   float* __restrict__ out_f32,
                   int mode, float scale) {
  const int lane = threadIdx.x & 31;
  const int w    = threadIdx.x >> 5;
  const int r    = lane & 15;   // A row within tile / B column within tile
  const int hh   = lane >> 4;   // lane half
  const int mbase = (blockIdx.x * 8 + w) * 32;   // 0..8160
  const int nbase = blockIdx.y * 64;             // 0..704

  const bf16* arow0 = A + (size_t)(mbase +      r) * EMBD;
  const bf16* arow1 = A + (size_t)(mbase + 16 + r) * EMBD;
  const bf16* wrow0 = W + (size_t)(nbase +  0 + r) * EMBD;
  const bf16* wrow1 = W + (size_t)(nbase + 16 + r) * EMBD;
  const bf16* wrow2 = W + (size_t)(nbase + 32 + r) * EMBD;
  const bf16* wrow3 = W + (size_t)(nbase + 48 + r) * EMBD;

  v8f acc[2][4] = {};
  for (int k0 = 0; k0 < EMBD; k0 += 32) {
    __builtin_prefetch(arow0 + k0 + 128, 0, 3);
    __builtin_prefetch(arow1 + k0 + 128, 0, 3);
    v16bf a0 = cat8(*(const v8bf*)(arow0 + k0 + hh * 8),
                    *(const v8bf*)(arow0 + k0 + 16 + hh * 8));
    v16bf a1 = cat8(*(const v8bf*)(arow1 + k0 + hh * 8),
                    *(const v8bf*)(arow1 + k0 + 16 + hh * 8));
    v16bf b0 = cat8(*(const v8bf*)(wrow0 + k0 + hh * 8), *(const v8bf*)(wrow0 + k0 + 16 + hh * 8));
    v16bf b1 = cat8(*(const v8bf*)(wrow1 + k0 + hh * 8), *(const v8bf*)(wrow1 + k0 + 16 + hh * 8));
    v16bf b2 = cat8(*(const v8bf*)(wrow2 + k0 + hh * 8), *(const v8bf*)(wrow2 + k0 + 16 + hh * 8));
    v16bf b3 = cat8(*(const v8bf*)(wrow3 + k0 + hh * 8), *(const v8bf*)(wrow3 + k0 + 16 + hh * 8));
    acc[0][0] = wmma_bf16(a0, b0, acc[0][0]);
    acc[1][0] = wmma_bf16(a1, b0, acc[1][0]);
    acc[0][1] = wmma_bf16(a0, b1, acc[0][1]);
    acc[1][1] = wmma_bf16(a1, b1, acc[1][1]);
    acc[0][2] = wmma_bf16(a0, b2, acc[0][2]);
    acc[1][2] = wmma_bf16(a1, b2, acc[1][2]);
    acc[0][3] = wmma_bf16(a0, b3, acc[0][3]);
    acc[1][3] = wmma_bf16(a1, b3, acc[1][3]);
  }

  #pragma unroll
  for (int mt = 0; mt < 2; ++mt) {
    #pragma unroll
    for (int nt = 0; nt < 4; ++nt) {
      const int n  = nbase + nt * 16 + r;            // output column 0..767
      const float bv = bias[n];
      #pragma unroll
      for (int v = 0; v < 8; ++v) {
        const int row = mbase + mt * 16 + hh * 8 + v; // output row 0..8191
        const float o = (acc[mt][nt][v] + bv) * scale;
        if (mode == 0) {
          const int b = row >> 10, t = row & 1023;
          const int h = n >> 6,   d = n & 63;
          out_bhtd[((size_t)((b * NHEAD + h) * SEQ + t)) * HDIM + d] = (bf16)o;
        } else {
          out_f32[(size_t)row * EMBD + n] = o;
        }
      }
    }
  }
}

// ---------------------------------------------------------------------------
// Flash attention. Q pre-scaled by 1/sqrt(D). Q,K,V bf16 [B,H,T,D].
// Block = 4 waves = 64 query rows; keys in chunks of 32 with online softmax.
// Writes Y as bf16 in [B,T,C] (token-major) for the output projection.
// ---------------------------------------------------------------------------
__global__ __launch_bounds__(128)
void csa_flash(const bf16* __restrict__ Q, const bf16* __restrict__ K,
               const bf16* __restrict__ V, bf16* __restrict__ Y) {
  __shared__ __align__(32) bf16 Vt[HDIM][32];     // transposed V chunk (shared)
  __shared__ __align__(32) bf16 Pst[4][16][32];   // per-wave P staging

  const int tid  = threadIdx.x;
  const int lane = tid & 31;
  const int w    = tid >> 5;
  const int r    = lane & 15;
  const int hh   = lane >> 4;
  const int bh   = blockIdx.y;                    // 0..95
  const int qw   = blockIdx.x * 64 + w * 16;      // first query row of wave

  const size_t base = (size_t)bh * SEQ * HDIM;
  const bf16* Qb = Q + base;
  const bf16* Kb = K + base;
  const bf16* Vb = V + base;

  // Q A-fragments for the two 32-wide D chunks
  const bf16* qrow = Qb + (size_t)(qw + r) * HDIM;
  const v16bf aQ0 = cat8(*(const v8bf*)(qrow +      hh * 8),
                         *(const v8bf*)(qrow + 16 + hh * 8));
  const v16bf aQ1 = cat8(*(const v8bf*)(qrow + 32 + hh * 8),
                         *(const v8bf*)(qrow + 48 + hh * 8));

  float m[8], ssum[8];
  v8f O[4] = {};
  #pragma unroll
  for (int v = 0; v < 8; ++v) { m[v] = -1e30f; ssum[v] = 0.0f; }

  const int kmax = blockIdx.x * 64 + 64;
  for (int j = 0; j < kmax; j += 32) {
    // Cooperative transpose-load V[j..j+32) x 64 -> Vt[d][k]
    {
      const int vr = tid >> 2;            // key row 0..31
      const int c0 = (tid & 3) * 16;      // d base
      const bf16* vrow = Vb + (size_t)(j + vr) * HDIM + c0;
      #pragma unroll
      for (int e = 0; e < 16; ++e) Vt[c0 + e][vr] = vrow[e];
    }
    __syncthreads();

    if (j <= qw + 15) {                   // wave has unmasked work this chunk
      // S = Q * K^T for two 16-key tiles
      const bf16* krow0 = Kb + (size_t)(j + r) * HDIM;
      const bf16* krow1 = Kb + (size_t)(j + 16 + r) * HDIM;
      v16bf bK00 = cat8(*(const v8bf*)(krow0 +      hh * 8), *(const v8bf*)(krow0 + 16 + hh * 8));
      v16bf bK01 = cat8(*(const v8bf*)(krow0 + 32 + hh * 8), *(const v8bf*)(krow0 + 48 + hh * 8));
      v16bf bK10 = cat8(*(const v8bf*)(krow1 +      hh * 8), *(const v8bf*)(krow1 + 16 + hh * 8));
      v16bf bK11 = cat8(*(const v8bf*)(krow1 + 32 + hh * 8), *(const v8bf*)(krow1 + 48 + hh * 8));
      v8f s0 = {}, s1 = {};
      s0 = wmma_bf16(aQ0, bK00, s0);
      s0 = wmma_bf16(aQ1, bK01, s0);
      s1 = wmma_bf16(aQ0, bK10, s1);
      s1 = wmma_bf16(aQ1, bK11, s1);

      float p0[8], p1[8], sc[8];
      #pragma unroll
      for (int v = 0; v < 8; ++v) {
        const int rowg = qw + hh * 8 + v;
        const float e0 = (j + r      <= rowg) ? s0[v] : -1e30f;  // causal mask
        const float e1 = (j + 16 + r <= rowg) ? s1[v] : -1e30f;
        // row max across the 16 lanes holding this row's columns
        float t = fmaxf(e0, e1);
        t = fmaxf(t, __shfl_xor(t, 1));
        t = fmaxf(t, __shfl_xor(t, 2));
        t = fmaxf(t, __shfl_xor(t, 4));
        t = fmaxf(t, __shfl_xor(t, 8));
        const float mn = fmaxf(m[v], t);
        const float rescale = __expf(m[v] - mn);
        const float q0 = __expf(e0 - mn);
        const float q1 = __expf(e1 - mn);
        float rs = q0 + q1;
        rs += __shfl_xor(rs, 1);
        rs += __shfl_xor(rs, 2);
        rs += __shfl_xor(rs, 4);
        rs += __shfl_xor(rs, 8);
        ssum[v] = ssum[v] * rescale + rs;
        m[v] = mn;
        p0[v] = q0; p1[v] = q1; sc[v] = rescale;
      }
      #pragma unroll
      for (int nt = 0; nt < 4; ++nt)
        #pragma unroll
        for (int v = 0; v < 8; ++v) O[nt][v] *= sc[v];

      // P (D-layout) -> LDS -> A-layout fragment
      #pragma unroll
      for (int v = 0; v < 8; ++v) {
        Pst[w][hh * 8 + v][r]      = (bf16)p0[v];
        Pst[w][hh * 8 + v][16 + r] = (bf16)p1[v];
      }
      asm volatile("s_wait_dscnt 0" ::: "memory");
      const v16bf aP = cat8(*(const v8bf*)&Pst[w][r][hh * 8],
                            *(const v8bf*)&Pst[w][r][16 + hh * 8]);
      #pragma unroll
      for (int nt = 0; nt < 4; ++nt) {
        const v16bf bV = cat8(*(const v8bf*)&Vt[nt * 16 + r][hh * 8], *(const v8bf*)&Vt[nt * 16 + r][16 + hh * 8]);
        O[nt] = wmma_bf16(aP, bV, O[nt]);
      }
    }
    __syncthreads();
  }

  // Epilogue: normalize, write token-major bf16
  const int b = bh / NHEAD, h = bh % NHEAD;
  #pragma unroll
  for (int v = 0; v < 8; ++v) {
    const int t = qw + hh * 8 + v;
    const float inv = 1.0f / ssum[v];
    #pragma unroll
    for (int nt = 0; nt < 4; ++nt) {
      const int c = h * HDIM + nt * 16 + r;
      Y[((size_t)(b * SEQ + t)) * EMBD + c] = (bf16)(O[nt][v] * inv);
    }
  }
}

// ---------------------------------------------------------------------------
// Host launcher
// ---------------------------------------------------------------------------
extern "C" void kernel_launch(void* const* d_in, const int* in_sizes, int n_in,
                              void* d_out, int out_size, void* d_ws, size_t ws_size,
                              hipStream_t stream) {
  (void)in_sizes; (void)n_in; (void)out_size; (void)ws_size;
  const float* x  = (const float*)d_in[0];
  const float* Wq = (const float*)d_in[1];
  const float* bq = (const float*)d_in[2];
  const float* Wk = (const float*)d_in[3];
  const float* bk = (const float*)d_in[4];
  const float* Wv = (const float*)d_in[5];
  const float* bv = (const float*)d_in[6];
  const float* Wo = (const float*)d_in[7];
  const float* bo = (const float*)d_in[8];

  char* p = (char*)d_ws;
  const size_t szX = (size_t)BT * EMBD * sizeof(bf16);     // 12.6 MB
  const size_t szW = (size_t)EMBD * EMBD * sizeof(bf16);   // 1.2 MB
  bf16* xb  = (bf16*)p; p += szX;
  bf16* wqb = (bf16*)p; p += szW;
  bf16* wkb = (bf16*)p; p += szW;
  bf16* wvb = (bf16*)p; p += szW;
  bf16* wob = (bf16*)p; p += szW;
  bf16* qb  = (bf16*)p; p += szX;
  bf16* kb  = (bf16*)p; p += szX;
  bf16* vb  = (bf16*)p; p += szX;
  bf16* yb  = (bf16*)p; p += szX;

  const int nX = BT * EMBD, nW = EMBD * EMBD;
  csa_cvt_bf16<<<1024, 256, 0, stream>>>(x,  xb,  nX);
  csa_cvt_bf16<<<256,  256, 0, stream>>>(Wq, wqb, nW);
  csa_cvt_bf16<<<256,  256, 0, stream>>>(Wk, wkb, nW);
  csa_cvt_bf16<<<256,  256, 0, stream>>>(Wv, wvb, nW);
  csa_cvt_bf16<<<256,  256, 0, stream>>>(Wo, wob, nW);

  const dim3 gg(BT / 256, EMBD / 64);   // (32, 12): 256 M-tiles x 12 N-strips
  // Q pre-scaled by 1/sqrt(HDIM) = 0.125
  csa_gemm_bf16<<<gg, 256, 0, stream>>>(xb, wqb, bq, qb, nullptr, 0, 0.125f);
  csa_gemm_bf16<<<gg, 256, 0, stream>>>(xb, wkb, bk, kb, nullptr, 0, 1.0f);
  csa_gemm_bf16<<<gg, 256, 0, stream>>>(xb, wvb, bv, vb, nullptr, 0, 1.0f);

  csa_flash<<<dim3(SEQ / 64, BATCH * NHEAD), 128, 0, stream>>>(qb, kb, vb, yb);

  csa_gemm_bf16<<<gg, 256, 0, stream>>>(yb, wob, bo, nullptr, (float*)d_out, 1, 1.0f);
}
